// repuxnet_conv_27230092657297
// MI455X (gfx1250) — matrix operand order repaired and checked by reading, not compiled, not run on hardware
//
#include <hip/hip_runtime.h>
#include <hip/hip_bf16.h>

// ---------------------------------------------------------------------------
// RepUX-Net forward for MI455X (gfx1250).
// Dominant cost: 21^3 depthwise convs (~261 GFLOP, L2-resident) -> routed
// through v_wmma_f32_16x16x32_f16 via a banded-matrix GEMM along W.
// B operands are pre-swizzled into parity-padded rows in global memory
// (contiguous 32B per lane, VMEM pipe); A operands stream from an LDS slab
// (DS pipe); each B is reused across NT output d-planes.
// ---------------------------------------------------------------------------

typedef __attribute__((ext_vector_type(16))) _Float16 v16h;
typedef __attribute__((ext_vector_type(8)))  float    v8f;

union V16 { v16h h; unsigned int u[8]; _Float16 e[16]; };
union V8  { v8f  v; float f[8]; };

__device__ __forceinline__ float gelu_exact(float x) {
    return 0.5f * x * (1.0f + erff(x * 0.70710678118654752f));
}

// ---------------------------------------------------------------------------
// Weight prep: effective taps wt = dw_w * BFM (alpha=1), stored as TWO
// parity-shifted zero-padded f16 copies per (c,kd,kh) row:
//   copy p (p=0: lead L=15, p=1: lead L=16), pitch 48:
//   wpad[((c*441 + kd*21+kh)*2 + p)*48 + i] = (0<=i-L<=20) ? wt[i-L] : 0
// A lane with column n, half hf reads 16 contiguous f16 starting at
// element t0+L (t0 = 16*hf - n), choosing L so the offset is even.
// ---------------------------------------------------------------------------
__global__ void wprep_kernel(const float* __restrict__ dww,
                             _Float16* __restrict__ wpad, int C) {
    int i = blockIdx.x * blockDim.x + threadIdx.x;
    int total = C * 441 * 96;
    if (i >= total) return;
    int e  = i % 96;         // element within (row, both copies)
    int r  = (i / 96) % 441; // kd*21+kh
    int c  = i / (96 * 441);
    int p  = e / 48;         // parity copy
    int ii = e % 48;
    int L  = 15 + p;
    int t  = ii - L;
    _Float16 v = (_Float16)0.0f;
    if (t >= 0 && t <= 20) {
        int kh = r % 21, kd = r / 21;
        float di = (float)(kd - 10), dj = (float)(kh - 10), dk = (float)(t - 10);
        float dist = sqrtf(di * di + dj * dj + dk * dk);
        float bfm  = 1.0f / (dist + 1.0f);
        v = (_Float16)(dww[(c * 441 + r) * 21 + t] * bfm);
    }
    wpad[i] = v;
}

// ---------------------------------------------------------------------------
// f32 -> f16 conversion of the activation tensor (conv input).
// ---------------------------------------------------------------------------
__global__ void cvt_kernel(const float* __restrict__ in,
                           _Float16* __restrict__ out, int n) {
    int i = blockIdx.x * blockDim.x + threadIdx.x;
    if (i < n) out[i] = (_Float16)in[i];
}

// ---------------------------------------------------------------------------
// Stem: Conv3d(1->48, k=7, s=2, p=3) fused with channels-first LayerNorm.
// ---------------------------------------------------------------------------
__global__ void stem_kernel(const float* __restrict__ x,
                            const float* __restrict__ w,
                            const float* __restrict__ b,
                            const float* __restrict__ lnw,
                            const float* __restrict__ lnb,
                            float* __restrict__ out) {
    extern __shared__ char smem_raw[];
    float* swgt = (float*)smem_raw;           // 48*343
    float* sb   = swgt + 48 * 343;
    float* slw  = sb + 48;
    float* slb  = slw + 48;
    for (int i = threadIdx.x; i < 48 * 343; i += blockDim.x) swgt[i] = w[i];
    for (int i = threadIdx.x; i < 48; i += blockDim.x) {
        sb[i] = b[i]; slw[i] = lnw[i]; slb[i] = lnb[i];
    }
    __syncthreads();

    int v = blockIdx.x * blockDim.x + threadIdx.x;
    const int S3 = 48 * 48 * 48;
    if (v >= S3) return;
    int ow = v % 48, t = v / 48;
    int oh = t % 48, od = t / 48;

    float acc[48];
#pragma unroll
    for (int c = 0; c < 48; ++c) acc[c] = sb[c];

    for (int kd = 0; kd < 7; ++kd) {
        int id = od * 2 - 3 + kd;
        if ((unsigned)id >= 96u) continue;
        for (int kh = 0; kh < 7; ++kh) {
            int ih = oh * 2 - 3 + kh;
            if ((unsigned)ih >= 96u) continue;
            for (int kw = 0; kw < 7; ++kw) {
                int iw = ow * 2 - 3 + kw;
                if ((unsigned)iw >= 96u) continue;
                float xv = x[(id * 96 + ih) * 96 + iw];
                const float* wp = swgt + (kd * 7 + kh) * 7 + kw;
#pragma unroll
                for (int c = 0; c < 48; ++c)
                    acc[c] = fmaf(wp[c * 343], xv, acc[c]);
            }
        }
    }
    float mean = 0.f;
#pragma unroll
    for (int c = 0; c < 48; ++c) mean += acc[c];
    mean *= (1.0f / 48.0f);
    float var = 0.f;
#pragma unroll
    for (int c = 0; c < 48; ++c) { float d = acc[c] - mean; var += d * d; }
    var *= (1.0f / 48.0f);
    float inv = rsqrtf(var + 1e-6f);
#pragma unroll
    for (int c = 0; c < 48; ++c)
        out[c * S3 + v] = (acc[c] - mean) * inv * slw[c] + slb[c];
}

// ---------------------------------------------------------------------------
// Depthwise 21^3 conv + folded BatchNorm + exact GELU via WMMA.
// Workgroup = (channel c, NT output d-planes, 16-row h block); 4 waves,
// wave w owns output columns [12w, 12w+12).
// Per (kd,kh): ONE banded-B load (8 global dwords/lane, contiguous) reused
// for NT WMMAs whose A rows stream from the LDS input slab.
// ---------------------------------------------------------------------------
template <int NT>
__global__ __launch_bounds__(128)
void dw_wmma_kernel(const _Float16* __restrict__ xin,
                    const _Float16* __restrict__ wpad,
                    const float* __restrict__ dwb,
                    const float* __restrict__ bng,
                    const float* __restrict__ bnb,
                    const float* __restrict__ bnm,
                    const float* __restrict__ bnv,
                    float* __restrict__ out,
                    int S, int Wlds, int nW) {
    extern __shared__ char smem_raw[];
    _Float16* inlds = (_Float16*)smem_raw;      // (20+NT) * 36 * Wlds

    const int c  = blockIdx.x;
    const int d0 = blockIdx.y * NT;
    const int h0 = blockIdx.z * 16;
    const int S2 = S * S, S3 = S2 * S;
    const int NPLANES = 20 + NT;

    // ---- stage input slab into LDS (zero padded halo, f16) ----
    const _Float16* xc = xin + (long)c * S3;
    const int tot = NPLANES * 36 * Wlds;
    for (int idx = threadIdx.x; idx < tot; idx += blockDim.x) {
        int iw = idx % Wlds;
        int t2 = idx / Wlds;
        int ih = t2 % 36;
        int p  = t2 / 36;
        int d = d0 + p - 10, h = h0 + ih - 10, w = iw - 10;
        _Float16 v = (_Float16)0.0f;
        if ((unsigned)d < (unsigned)S && (unsigned)h < (unsigned)S &&
            (unsigned)w < (unsigned)S)
            v = xc[d * S2 + h * S + w];
        inlds[idx] = v;
    }
    __syncthreads();

    const int wave = threadIdx.x >> 5;
    if (wave >= nW) return;                 // wave-uniform: EXEC stays all-1s
    const int lane = threadIdx.x & 31;
    const int n  = lane & 15;               // B column / A row / D column
    const int hf = lane >> 4;               // lane half
    const int wb = wave * 12;               // output w tile base

    // Per-lane banded-B base: contiguous 16 taps starting at t0=16*hf-n,
    // taken from the parity copy that makes the element offset even.
    const int t0 = 16 * hf - n;
    const int base_el = (n & 1) ? (t0 + 15) : (48 + t0 + 16);
    const _Float16* wrow = wpad + (long)c * 441 * 96 + base_el;

    V8 acc[NT];
#pragma unroll
    for (int i = 0; i < NT; ++i)
#pragma unroll
        for (int r = 0; r < 8; ++r) acc[i].f[r] = 0.f;

    for (int kd = 0; kd < 21; ++kd) {
        for (int kh = 0; kh < 21; ++kh) {
            // ---- B: 8 contiguous dwords from global (L2 resident) ----
            const unsigned int* bp = reinterpret_cast<const unsigned int*>(
                wrow + (kd * 21 + kh) * 96);
            V16 bm;
#pragma unroll
            for (int v = 0; v < 8; ++v) bm.u[v] = bp[v];

            // ---- NT A tiles from LDS, one WMMA each ----
            const _Float16* abase =
                inlds + (kd * 36 + (n + kh)) * Wlds + wb + 16 * hf;
#pragma unroll
            for (int i = 0; i < NT; ++i) {
                const _Float16* arow = abase + i * (36 * Wlds);
                V16 am;
#pragma unroll
                for (int v = 0; v < 8; ++v) {
                    int off = 2 * v; // K 0..15
                    am.u[v] = *reinterpret_cast<const unsigned int*>(arow + off);
                }
                acc[i].v = __builtin_amdgcn_wmma_f32_16x16x32_f16(
                    false, am.h, false, bm.h, (short)0, acc[i].v, false, false);
            }
        }
    }

    // ---- epilogue: folded bias+BN then exact GELU ----
    float scale = bng[c] * rsqrtf(bnv[c] + 1e-5f);
    float shift = (dwb[c] - bnm[c]) * scale + bnb[c];
#pragma unroll
    for (int i = 0; i < NT; ++i) {
        float* oc = out + (long)c * S3 + (d0 + i) * S2;
#pragma unroll
        for (int r = 0; r < 8; ++r) {
            int m  = 8 * hf + r;
            int oh = h0 + m, ow = wb + n;
            if (n < 12 && oh < S && ow < S) {
                float y = acc[i].f[r] * scale + shift;
                oc[oh * S + ow] = gelu_exact(y);
            }
        }
    }
}

// ---------------------------------------------------------------------------
// Channels-first LayerNorm: one thread per spatial voxel, reduce over C.
// ---------------------------------------------------------------------------
__global__ void ln_kernel(const float* __restrict__ in,
                          const float* __restrict__ w,
                          const float* __restrict__ b,
                          float* __restrict__ out, int C, int S3) {
    int v = blockIdx.x * blockDim.x + threadIdx.x;
    if (v >= S3) return;
    float s = 0.f, s2 = 0.f;
    for (int c = 0; c < C; ++c) {
        float x = in[c * S3 + v];
        s += x; s2 += x * x;
    }
    float mean = s / (float)C;
    float var  = s2 / (float)C - mean * mean;
    float inv  = rsqrtf(var + 1e-6f);
    for (int c = 0; c < C; ++c) {
        float x = in[c * S3 + v];
        out[c * S3 + v] = (x - mean) * inv * w[c] + b[c];
    }
}

// ---------------------------------------------------------------------------
// Downsample Conv3d(Ci->Co, k=2, s=2, p=0).
// ---------------------------------------------------------------------------
__global__ void ds_kernel(const float* __restrict__ in,
                          const float* __restrict__ w,
                          const float* __restrict__ b,
                          float* __restrict__ out,
                          int Ci, int Si, int Co) {
    int So = Si >> 1;
    int So3 = So * So * So;
    long gid = (long)blockIdx.x * blockDim.x + threadIdx.x;
    if (gid >= (long)Co * So3) return;
    int v  = (int)(gid % So3);
    int co = (int)(gid / So3);
    int ow = v % So, t = v / So;
    int oh = t % So, od = t / So;
    int Si2 = Si * Si;
    const float* wc = w + (long)co * Ci * 8;
    float acc = b[co];
    for (int ci = 0; ci < Ci; ++ci) {
        const float* xp = in + (long)ci * Si * Si2 + (2 * od) * Si2 +
                          (2 * oh) * Si + 2 * ow;
        const float* wp = wc + ci * 8;
        acc += wp[0] * xp[0]           + wp[1] * xp[1]
             + wp[2] * xp[Si]          + wp[3] * xp[Si + 1]
             + wp[4] * xp[Si2]         + wp[5] * xp[Si2 + 1]
             + wp[6] * xp[Si2 + Si]    + wp[7] * xp[Si2 + Si + 1];
    }
    out[gid] = acc;
}

// ---------------------------------------------------------------------------
// Host side: input resolution + launch sequence.
// ---------------------------------------------------------------------------
struct Net {
    const float *x, *stem_w, *stem_b, *stem_lnw, *stem_lnb;
    const float *ds_lw[3], *ds_lb[3], *ds_cw[3], *ds_cb[3];
    const float *dw_w[4][2], *dw_b[4][2];
    const float *bn_g[4][2], *bn_b[4][2], *bn_m[4][2], *bn_v[4][2];
    const float *nw[4], *nb[4];
};

static void resolve(Net* P, void* const* din, const int* sz, int n_in) {
    auto F = [&](int i) { return (const float*)din[i]; };
    if (n_in > 0 && sz[0] == 884736) {
        // insertion-order flatten: x, then params in declaration order
        P->x = F(0); P->stem_w = F(1); P->stem_b = F(2);
        P->stem_lnw = F(3); P->stem_lnb = F(4);
        int k = 5;
        for (int s = 0; s < 3; ++s) {
            P->ds_lw[s] = F(k++); P->ds_lb[s] = F(k++);
            P->ds_cw[s] = F(k++); P->ds_cb[s] = F(k++);
        }
        for (int st = 0; st < 4; ++st)
            for (int b = 0; b < 2; ++b) {
                P->dw_w[st][b] = F(k++); P->dw_b[st][b] = F(k++);
                P->bn_g[st][b] = F(k++); P->bn_b[st][b] = F(k++);
                P->bn_m[st][b] = F(k++); P->bn_v[st][b] = F(k++);
            }
        for (int i = 0; i < 4; ++i) P->nw[i] = F(k++);
        for (int i = 0; i < 4; ++i) P->nb[i] = F(k++);
    } else {
        // jax tree_flatten order: dict keys sorted; 'params' < 'x'
        int k = 0;
        for (int s = 0; s < 3; ++s) {   // ds: conv_b, conv_w, ln_b, ln_w
            P->ds_cb[s] = F(k++); P->ds_cw[s] = F(k++);
            P->ds_lb[s] = F(k++); P->ds_lw[s] = F(k++);
        }
        for (int i = 0; i < 4; ++i) P->nb[i] = F(k++);   // norm_b
        for (int i = 0; i < 4; ++i) P->nw[i] = F(k++);   // norm_w
        for (int st = 0; st < 4; ++st)                   // stages
            for (int b = 0; b < 2; ++b) {                // bn_b,bn_g,bn_m,bn_v,dw_b,dw_w
                P->bn_b[st][b] = F(k++); P->bn_g[st][b] = F(k++);
                P->bn_m[st][b] = F(k++); P->bn_v[st][b] = F(k++);
                P->dw_b[st][b] = F(k++); P->dw_w[st][b] = F(k++);
            }
        P->stem_b = F(k++); P->stem_w = F(k++);
        P->stem_lnb = F(k++); P->stem_lnw = F(k++);
        P->x = F(k++);
    }
}

extern "C" void kernel_launch(void* const* d_in, const int* in_sizes, int n_in,
                              void* d_out, int out_size, void* d_ws, size_t ws_size,
                              hipStream_t stream) {
    (void)out_size; (void)ws_size;
    Net P;
    resolve(&P, d_in, in_sizes, n_in);

    // Workspace layout (~82 MB): two f32 activation buffers, f16 input
    // staging, parity-padded f16 banded-weight buffer.
    char* base = (char*)d_ws;
    const size_t ACT_BYTES = 21241856;   // >= 48*48^3*4, 4KB aligned
    const size_t F16_BYTES = 10620928;   // >= 48*48^3*2
    float*    actA   = (float*)base;
    float*    actB   = (float*)(base + ACT_BYTES);
    _Float16* f16Buf = (_Float16*)(base + 2 * ACT_BYTES);
    _Float16* wpadBuf = (_Float16*)(base + 2 * ACT_BYTES + F16_BYTES);
    // wpad max: 384*441*96*2 B = 32,514,048

    static const int Sd[4] = {48, 24, 12, 6};
    static const int Cd[4] = {48, 96, 192, 384};

    // ---- stem conv + LN ----
    {
        size_t shmem = (size_t)(48 * 343 + 3 * 48) * sizeof(float);
        int S3 = 48 * 48 * 48;
        stem_kernel<<<(S3 + 255) / 256, 256, shmem, stream>>>(
            P.x, P.stem_w, P.stem_b, P.stem_lnw, P.stem_lnb, actA);
    }

    float* cur = actA;
    float* nxt = actB;
    size_t outOff = 0;

    for (int st = 0; st < 4; ++st) {
        int C = Cd[st], S = Sd[st];
        int S3 = S * S * S;
        int Wlds = (S + 20 < 32) ? 32 : (S + 20);
        int nW = (S + 11) / 12;
        int NT = (S % 4 == 0) ? 4 : 2;
        size_t shmem = (size_t)((20 + NT) * 36 * Wlds) * sizeof(_Float16);

        for (int b = 0; b < 2; ++b) {
            wprep_kernel<<<(C * 441 * 96 + 255) / 256, 256, 0, stream>>>(
                P.dw_w[st][b], wpadBuf, C);
            cvt_kernel<<<(C * S3 + 255) / 256, 256, 0, stream>>>(
                cur, f16Buf, C * S3);
            dim3 grid(C, S / NT, (S + 15) / 16);
            if (NT == 4)
                dw_wmma_kernel<4><<<grid, 128, shmem, stream>>>(
                    f16Buf, wpadBuf, P.dw_b[st][b], P.bn_g[st][b],
                    P.bn_b[st][b], P.bn_m[st][b], P.bn_v[st][b], nxt,
                    S, Wlds, nW);
            else
                dw_wmma_kernel<2><<<grid, 128, shmem, stream>>>(
                    f16Buf, wpadBuf, P.dw_b[st][b], P.bn_g[st][b],
                    P.bn_b[st][b], P.bn_m[st][b], P.bn_v[st][b], nxt,
                    S, Wlds, nW);
            float* tmp = cur; cur = nxt; nxt = tmp;
        }

        // stage output: LN -> d_out segment
        ln_kernel<<<(S3 + 255) / 256, 256, 0, stream>>>(
            cur, P.nw[st], P.nb[st], (float*)d_out + outOff, C, S3);
        outOff += (size_t)C * S3;

        if (st < 3) {
            // downsample: LN (cur -> nxt), then 2^3/s2 conv (nxt -> cur)
            ln_kernel<<<(S3 + 255) / 256, 256, 0, stream>>>(
                cur, P.ds_lw[st], P.ds_lb[st], nxt, C, S3);
            int C2 = Cd[st + 1];
            int So = S / 2;
            long tot = (long)C2 * So * So * So;
            ds_kernel<<<(int)((tot + 255) / 256), 256, 0, stream>>>(
                nxt, P.ds_cw[st], P.ds_cb[st], cur, C, S, C2);
        }
    }
}
